// WeightOnlyInt4Linear_7387343749643
// MI455X (gfx1250) — matrix operand order repaired and checked by reading, not compiled, not run on hardware
//
#include <hip/hip_runtime.h>
#include <hip/hip_bf16.h>

// ---- types ------------------------------------------------------------
typedef __bf16 bf16_t;
typedef __attribute__((ext_vector_type(16))) __bf16 v16bf;
typedef __attribute__((ext_vector_type(8)))  __bf16 v8bf;
typedef __attribute__((ext_vector_type(4)))  __bf16 v4bf;
typedef __attribute__((ext_vector_type(8)))  float  v8f;

#define K_DIM 4096
#define N_DIM 11008
#define NGRP  32          // 4096 / 128 groups along K
#define BM    128
#define BN    128
#define BK    64
#define LDA   (BK + 8)    // padded LDS row stride (144 B, 16-B aligned)

// ---------------------------------------------------------------------
// W4A32 GEMM: out[M,N] = x[M,K] * dequant(packed_weight)[K,N]
// bf16 WMMA with f32 accumulation. Block: 256 thr = 8 waves,
// tile 128x128, K-step 64 (2 x k=32 WMMA), outer loop over quant groups.
// ---------------------------------------------------------------------
__global__ __launch_bounds__(256, 2)
void w4_gemm_bf16_wmma(const float* __restrict__ x,
                       const int*   __restrict__ pw,
                       const float* __restrict__ scales,
                       const float* __restrict__ zeros,
                       float* __restrict__ out,
                       int M) {
  __shared__ __align__(16) bf16_t As[BM][LDA];   // [m][k]  bf16 activations
  __shared__ __align__(16) bf16_t Bs[BN][LDA];   // [n][k]  bf16 dequant weights

  const int tid  = threadIdx.x;
  const int lane = tid & 31;
  const int wave = tid >> 5;
  const int wm   = wave & 3;    // 4 waves along M -> 32-row strips
  const int wn   = wave >> 2;   // 2 waves along N -> 64-col strips

  const int m0 = blockIdx.y * BM;
  const int n0 = blockIdx.x * BN;

  v8f acc[2][4];
  #pragma unroll
  for (int i = 0; i < 2; ++i)
    #pragma unroll
    for (int j = 0; j < 4; ++j)
      acc[i][j] = v8f{};

  // B-tile loader mapping: 1 word-row of 4 consecutive columns per thread
  const int jw = tid >> 5;      // 0..7 packed-word row within the K-block
  const int nb = (tid & 31) * 4;// base column within tile (4 cols/thread)

  const int half = lane >> 4;
  const int l15  = lane & 15;

  // Outer loop over quant groups (g fixed for 2 K-steps of 64)
  for (int g = 0; g < NGRP; ++g) {
    // scales/zeros for this thread's 4 columns, hoisted out of the K-steps
    float sc[4], fz[4];
    #pragma unroll
    for (int c = 0; c < 4; ++c) {
      const int n = n0 + nb + c;
      const float s = scales[(size_t)n * NGRP + g];
      const float z = zeros [(size_t)n * NGRP + g];
      sc[c] = s;
      fz[c] = -z * s;            // dequant becomes one FMA: q*s + (-z*s)
    }

    #pragma unroll
    for (int kk = 0; kk < 2; ++kk) {
      const int k0 = g * 128 + kk * BK;

      // ---------------- stage A: 128x64 fp32 -> bf16 in LDS ----------------
      #pragma unroll
      for (int i = 0; i < 8; ++i) {
        const int f   = tid + i * 256;      // float4 index, 2048 total
        const int row = f >> 4;
        const int c4  = (f & 15) << 2;
        const float4 v = *(const float4*)(x + (size_t)(m0 + row) * K_DIM + k0 + c4);
        v4bf p;
        p.x = (bf16_t)v.x; p.y = (bf16_t)v.y; p.z = (bf16_t)v.z; p.w = (bf16_t)v.w;
        *(v4bf*)&As[row][c4] = p;           // 8-byte LDS store
      }

      // ------ stage B: one coalesced b128 of packed words, dequant ------
      {
        const int kw = (k0 >> 3) + jw;      // packed word row
        const int4 wq = *(const int4*)(pw + (size_t)kw * N_DIM + n0 + nb);
        const int wv[4] = {wq.x, wq.y, wq.z, wq.w};
        #pragma unroll
        for (int c = 0; c < 4; ++c) {
          const int w = wv[c];
          v8bf qv;
          #pragma unroll
          for (int nib = 0; nib < 8; ++nib) {
            const float q = (float)((w >> (nib * 4)) & 0xF);
            qv[nib] = (bf16_t)fmaf(q, sc[c], fz[c]);
          }
          *(v8bf*)&Bs[nb + c][jw * 8] = qv; // 16-byte LDS store
        }
      }

      // prefetch next K-tile (gfx1250 global_prefetch path)
      if (k0 + BK < K_DIM) {
        __builtin_prefetch(x  + (size_t)(m0 + (tid >> 1)) * K_DIM + (k0 + BK), 0, 0);
        __builtin_prefetch(pw + (size_t)(((k0 + BK) >> 3) + jw) * N_DIM + n0 + nb, 0, 0);
      }

      __syncthreads();

      // ---------------- compute: 2 k-substeps of 32 ----------------
      #pragma unroll
      for (int ks = 0; ks < BK; ks += 32) {
        // A fragments: lane m holds K {0-7,16-23} (lanes 0-15) / {8-15,24-31}
        v16bf a[2];
        #pragma unroll
        for (int mt = 0; mt < 2; ++mt) {
          const bf16_t* ap = &As[wm * 32 + mt * 16 + l15][ks + half * 8];
          const v8bf lo = *(const v8bf*)ap;         // K +0..7
          const v8bf hi = *(const v8bf*)(ap + 16);  // K +16..23
          a[mt] = __builtin_shufflevector(lo, hi, 0,1,2,3,4,5,6,7,
                                                  8,9,10,11,12,13,14,15);
        }
        // B fragments: lane n holds K {0-7,16-23} (lanes 0-15) / {8-15,24-31} of column n
        v16bf b[4];
        #pragma unroll
        for (int nt = 0; nt < 4; ++nt) {
          const bf16_t* bp = &Bs[wn * 64 + nt * 16 + l15][ks + half * 8];
          const v8bf lo = *(const v8bf*)bp;
          const v8bf hi = *(const v8bf*)(bp + 16);
          b[nt] = __builtin_shufflevector(lo, hi, 0,1,2,3,4,5,6,7,
                                                  8,9,10,11,12,13,14,15);
        }
        // 8 WMMAs per substep, 16 per K-iteration
        #pragma unroll
        for (int mt = 0; mt < 2; ++mt)
          #pragma unroll
          for (int nt = 0; nt < 4; ++nt)
            acc[mt][nt] = __builtin_amdgcn_wmma_f32_16x16x32_bf16(
                /*neg_a=*/false, a[mt], /*neg_b=*/false, b[nt],
                /*c_mod=*/(short)0, acc[mt][nt],
                /*reuse_a=*/false, /*reuse_b=*/false);
      }

      __syncthreads();
    }
  }

  // ---------------- epilogue: scatter C per ISA layout ----------------
  // VGPR r: lanes 0-15 -> M=r, N=lane; lanes 16-31 -> M=r+8, N=lane-16
  #pragma unroll
  for (int mt = 0; mt < 2; ++mt) {
    #pragma unroll
    for (int nt = 0; nt < 4; ++nt) {
      const int col = n0 + wn * 64 + nt * 16 + l15;
      #pragma unroll
      for (int r = 0; r < 8; ++r) {
        const int row = m0 + wm * 32 + mt * 16 + r + half * 8;
        out[(size_t)row * N_DIM + col] = acc[mt][nt][r];
      }
    }
  }
}

// ---------------------------------------------------------------------
extern "C" void kernel_launch(void* const* d_in, const int* in_sizes, int n_in,
                              void* d_out, int out_size, void* d_ws, size_t ws_size,
                              hipStream_t stream) {
  const float* x      = (const float*)d_in[0];
  const int*   pw     = (const int*)  d_in[1];
  const float* scales = (const float*)d_in[2];
  const float* zeros  = (const float*)d_in[3];
  // d_in[4] = groupsize scalar (128), baked into the kernel constants.
  float* out = (float*)d_out;

  const int M = in_sizes[0] / K_DIM;          // 4*2048 = 8192
  dim3 grid(N_DIM / BN, M / BM);              // (86, 64)
  w4_gemm_bf16_wmma<<<grid, 256, 0, stream>>>(x, pw, scales, zeros, out, M);
}
